// _QuantizedLinear_24859270709405
// MI455X (gfx1250) — matrix operand order repaired and checked by reading, not compiled, not run on hardware
//
#include <hip/hip_runtime.h>
#include <hip/hip_fp16.h>

typedef __attribute__((ext_vector_type(16))) _Float16 v16h;
typedef __attribute__((ext_vector_type(8)))  float    v8f;

// Problem dims (fixed by the reference: B=4, S=2048, IN=4096, OUT=11008)
constexpr int M_ = 4 * 2048;   // 8192 rows of x
constexpr int N_ = 11008;      // output channels
constexpr int K_ = 4096;       // reduction dim

// Tiling
constexpr int BM = 128;
constexpr int BN = 128;
constexpr int BK = 32;
constexpr int KT = K_ / BK;    // 128 K-tiles
constexpr int LDST = BK + 8;   // LDS pitch in halfs (even -> dword-aligned pairs)

__global__ __launch_bounds__(256)
void qlinear_wmma_f16(const float* __restrict__ x,
                      const int*   __restrict__ wq,
                      const __half* __restrict__ scale,
                      const __half* __restrict__ bias,
                      float* __restrict__ out) {
    // Ping-pong LDS buffers: 2 * (128*40 + 128*40) halfs = 40 KB of 320 KB WGP pool
    __shared__ _Float16 As[2][BM][LDST];  // x tile,  [m][k]
    __shared__ _Float16 Bs[2][BN][LDST];  // w tile,  [n][k]

    const int tid    = threadIdx.x;
    const int lane   = tid & 31;
    const int wave   = tid >> 5;     // 0..7
    const int wm     = wave & 3;     // wave row  (4)
    const int wn     = wave >> 2;    // wave col  (2)
    const int lane16 = lane & 15;
    const int laneHi = lane >> 4;    // 0/1

    const int blockN0 = blockIdx.x * BN;
    const int blockM0 = blockIdx.y * BM;

    v8f acc[2][4];
    #pragma unroll
    for (int i = 0; i < 2; ++i)
        #pragma unroll
        for (int j = 0; j < 4; ++j) {
            v8f z = {};
            acc[i][j] = z;
        }

    // Per-thread global load coordinates: tile = 128 rows x 8 float4 columns
    // = 1024 float4 chunks; 256 threads -> 4 chunks each, same for A and B.
    int rowc[4], colc[4];
    #pragma unroll
    for (int i = 0; i < 4; ++i) {
        const int linear = i * 256 + tid;   // 0..1023
        rowc[i] = linear >> 3;              // 0..127
        colc[i] = (linear & 7) << 2;        // 0,4,...,28
    }

    float4 xa[4];
    int4   wa[4];

    // ---- prologue: fetch tile 0 and stage into LDS buffer 0 ----
    #pragma unroll
    for (int i = 0; i < 4; ++i) {
        xa[i] = *reinterpret_cast<const float4*>(
            &x [(size_t)(blockM0 + rowc[i]) * K_ + colc[i]]);
        wa[i] = *reinterpret_cast<const int4*>(
            &wq[(size_t)(blockN0 + rowc[i]) * K_ + colc[i]]);
    }
    #pragma unroll
    for (int i = 0; i < 4; ++i) {
        As[0][rowc[i]][colc[i] + 0] = (_Float16)xa[i].x;
        As[0][rowc[i]][colc[i] + 1] = (_Float16)xa[i].y;
        As[0][rowc[i]][colc[i] + 2] = (_Float16)xa[i].z;
        As[0][rowc[i]][colc[i] + 3] = (_Float16)xa[i].w;
        Bs[0][rowc[i]][colc[i] + 0] = (_Float16)(float)wa[i].x;  // exact: |w| <= 127
        Bs[0][rowc[i]][colc[i] + 1] = (_Float16)(float)wa[i].y;
        Bs[0][rowc[i]][colc[i] + 2] = (_Float16)(float)wa[i].z;
        Bs[0][rowc[i]][colc[i] + 3] = (_Float16)(float)wa[i].w;
    }
    __syncthreads();

    // ---- software-pipelined main loop ----
    for (int kt = 0; kt < KT; ++kt) {
        const int cur = kt & 1;
        const int nxt = cur ^ 1;

        // Issue next tile's global loads first; they stay in flight while the
        // WMMAs below consume the current LDS buffer (loadcnt waited only at
        // the cvt/store phase after the barrier).
        if (kt + 1 < KT) {
            const int k0n = (kt + 1) * BK;
            #pragma unroll
            for (int i = 0; i < 4; ++i) {
                xa[i] = *reinterpret_cast<const float4*>(
                    &x [(size_t)(blockM0 + rowc[i]) * K_ + k0n + colc[i]]);
                wa[i] = *reinterpret_cast<const int4*>(
                    &wq[(size_t)(blockN0 + rowc[i]) * K_ + k0n + colc[i]]);
            }
        }
        // Run-ahead prefetch two tiles out (gfx1250 global_prefetch_b8 path).
        if (kt + 2 < KT) {
            const int k0p = (kt + 2) * BK;
            __builtin_prefetch(&x [(size_t)(blockM0 + rowc[0]) * K_ + k0p + colc[0]], 0, 3);
            __builtin_prefetch(&wq[(size_t)(blockN0 + rowc[0]) * K_ + k0p + colc[0]], 0, 3);
        }

        // Build fragments from LDS (CDNA5 WMMA VGPR layouts).
        v16h afrag[2];
        #pragma unroll
        for (int i = 0; i < 2; ++i) {
            const int m = wm * 32 + i * 16 + lane16;
            union { unsigned u[8]; v16h h; } t;
            #pragma unroll
            for (int p = 0; p < 8; ++p) {
                const int h2 = p * 2;  // even half index
                const int k  = ((h2 < 8) ? 0 : 16) + (laneHi ? 8 : 0) + (h2 & 7);
                t.u[p] = *reinterpret_cast<const unsigned*>(&As[cur][m][k]);
            }
            afrag[i] = t.h;
        }
        v16h bfrag[4];
        #pragma unroll
        for (int j = 0; j < 4; ++j) {
            const int n = wn * 64 + j * 16 + lane16;
            union { unsigned u[8]; v16h h; } t;
            #pragma unroll
            for (int p = 0; p < 8; ++p) {
                const int k = ((p < 4) ? 0 : 16) + (laneHi ? 8 : 0) + ((p * 2) & 7);
                t.u[p] = *reinterpret_cast<const unsigned*>(&Bs[cur][n][k]);
            }
            bfrag[j] = t.h;
        }

        // 8 WMMAs per wave per K-step: D = A x B + C, fp32 accumulate.
        #pragma unroll
        for (int i = 0; i < 2; ++i)
            #pragma unroll
            for (int j = 0; j < 4; ++j)
                acc[i][j] = __builtin_amdgcn_wmma_f32_16x16x32_f16(
                    /*neg_a=*/false, afrag[i],
                    /*neg_b=*/false, bfrag[j],
                    /*c_mod=*/(short)0, acc[i][j],
                    /*reuse_a=*/false, /*reuse_b=*/false);

        __syncthreads();   // all reads of buffer `nxt` (from iter kt-1) done

        if (kt + 1 < KT) {
            #pragma unroll
            for (int i = 0; i < 4; ++i) {
                As[nxt][rowc[i]][colc[i] + 0] = (_Float16)xa[i].x;
                As[nxt][rowc[i]][colc[i] + 1] = (_Float16)xa[i].y;
                As[nxt][rowc[i]][colc[i] + 2] = (_Float16)xa[i].z;
                As[nxt][rowc[i]][colc[i] + 3] = (_Float16)xa[i].w;
                Bs[nxt][rowc[i]][colc[i] + 0] = (_Float16)(float)wa[i].x;
                Bs[nxt][rowc[i]][colc[i] + 1] = (_Float16)(float)wa[i].y;
                Bs[nxt][rowc[i]][colc[i] + 2] = (_Float16)(float)wa[i].z;
                Bs[nxt][rowc[i]][colc[i] + 3] = (_Float16)(float)wa[i].w;
            }
        }
        __syncthreads();   // buffer `nxt` ready for iter kt+1
    }

    // Epilogue: out[m,n] = acc * scale[n] + bias[n]  (fp32)
    // C layout: VGPR r -> M = r + 8*laneHi, lane%16 -> N.
    #pragma unroll
    for (int j = 0; j < 4; ++j) {
        const int n = blockN0 + wn * 64 + j * 16 + lane16;
        const float s = __half2float(scale[n]);
        const float b = __half2float(bias[n]);
        #pragma unroll
        for (int i = 0; i < 2; ++i) {
            const int mbase = blockM0 + wm * 32 + i * 16 + laneHi * 8;
            #pragma unroll
            for (int r = 0; r < 8; ++r) {
                out[(size_t)(mbase + r) * N_ + n] = acc[i][j][r] * s + b;
            }
        }
    }
}

extern "C" void kernel_launch(void* const* d_in, const int* in_sizes, int n_in,
                              void* d_out, int out_size, void* d_ws, size_t ws_size,
                              hipStream_t stream) {
    const float*  x     = (const float*)d_in[0];
    const int*    wq    = (const int*)d_in[1];
    const __half* scale = (const __half*)d_in[2];
    const __half* bias  = (const __half*)d_in[3];
    float*        out   = (float*)d_out;

    dim3 grid(N_ / BN, M_ / BM);   // 86 x 64 workgroups
    dim3 block(256);               // 8 waves (wave32)
    qlinear_wmma_f16<<<grid, block, 0, stream>>>(x, wq, scale, bias, out);
}
